// EdgeResidualLayer_73667279061346
// MI455X (gfx1250) — matrix operand order repaired and checked by reading, not compiled, not run on hardware
//
#include <hip/hip_runtime.h>

// ---------------- CDNA5 types ----------------
typedef __attribute__((ext_vector_type(16))) __bf16 v16bf;
typedef __attribute__((ext_vector_type(2)))  __bf16 bf16x2;
typedef __attribute__((ext_vector_type(8)))  float  v8f;
typedef __attribute__((ext_vector_type(4)))  int    v4i;

#define AS1 __attribute__((address_space(1)))
#define AS3 __attribute__((address_space(3)))

#define NODE_H 128
#define EDGE_H 128
#define IN_DIM 384
#define KT1 12            // 384 / 32 k-steps for GEMM1
#define KT2 4             // 128 / 32 k-steps for GEMM2
#define NT  8             // 128 / 16 n-tiles
#define W1_FRAG_DW (KT1 * NT * 256)   // 24576 dwords = 96 KB (bf16 B-fragments of W1)
#define W2_FRAG_DW (KT2 * NT * 256)   // 8192 dwords  = 32 KB (bf16 B-fragments of W2)

#if __has_builtin(__builtin_amdgcn_global_load_async_to_lds_b128) && \
    __has_builtin(__builtin_amdgcn_s_wait_asynccnt)
#define HAVE_ASYNC_LDS 1
#endif

// fp32 pair -> packed bf16 dword, round-to-nearest-even via native converts
__device__ __forceinline__ unsigned int pack_bf2(float lo, float hi) {
#if __has_builtin(__builtin_amdgcn_cvt_pk_bf16_f32)
    auto p = __builtin_amdgcn_cvt_pk_bf16_f32(lo, hi);   // v_cvt_pk_bf16_f32
    return __builtin_bit_cast(unsigned int, p);
#else
    bf16x2 p = { (__bf16)lo, (__bf16)hi };
    return __builtin_bit_cast(unsigned int, p);
#endif
}
__device__ __forceinline__ unsigned short f2bf_bits(float f) {
    return __builtin_bit_cast(unsigned short, (__bf16)f);
}

union Frag {
    unsigned int u[8];
    uint4 q[2];
    v16bf v;
};

// --------------------------------------------------------------------------
// Prep: convert W1 [384x128] and W2 [128x128] fp32 row-major into bf16
// B-matrix WMMA fragments (ISA 7.12.2: lane L = (K-half h, col n=L%16);
// VGPR j packs K = h*8 + 2j + (j>=4 ? 8 : 0) (lo 16b) and that + 1 (hi 16b)).
// ws dword layout: [frag = kt*8+nt][lane][j]
// --------------------------------------------------------------------------
__global__ void pack_weights_kernel(const float* __restrict__ W1,
                                    const float* __restrict__ W2,
                                    unsigned int* __restrict__ ws) {
    int tid = blockIdx.x * blockDim.x + threadIdx.x;
    if (tid >= W1_FRAG_DW + W2_FRAG_DW) return;
    const float* W;
    unsigned int* out;
    int t;
    if (tid < W1_FRAG_DW) { W = W1; out = ws;               t = tid; }
    else                  { W = W2; out = ws + W1_FRAG_DW;  t = tid - W1_FRAG_DW; }
    int f    = t >> 8;        // fragment index (kt*8 + nt)
    int rem  = t & 255;
    int lane = rem >> 3;
    int j    = rem & 7;
    int kt   = f >> 3;
    int nt   = f & 7;
    int h    = lane >> 4;
    int n    = lane & 15;
    int k0   = kt * 32 + h * 8 + 2 * j + (j >= 4 ? 8 : 0);
    int col  = nt * 16 + n;
    out[t] = pack_bf2(W[(size_t)k0 * EDGE_H + col], W[(size_t)(k0 + 1) * EDGE_H + col]);
}

// --------------------------------------------------------------------------
// Main: 256 threads = 8 waves, 128 edges per block.
// Wave w handles edges [blk*128 + 16w, +16): GEMM1 (K=384) -> LDS transpose
// -> GEMM2 (K=128) -> fp32 residual epilogue. All matrix math via
// v_wmma_f32_16x16x32_bf16 with fp32 accumulators.
// --------------------------------------------------------------------------
__global__ __launch_bounds__(256) void edge_mlp_kernel(
    const float* __restrict__ x,
    const int*   __restrict__ ei,   // [2][E] (int32)
    const float* __restrict__ ea,   // [E][128]
    const float* __restrict__ b1,
    const float* __restrict__ b2,
    const unsigned int* __restrict__ w1f,   // W1 bf16 fragments (global, L2-resident)
    const unsigned int* __restrict__ w2f,   // W2 bf16 fragments
    float* __restrict__ out,
    int E)
{
    __shared__ __align__(16) unsigned int  lds_w2[W2_FRAG_DW];        // 32 KB
    __shared__ __align__(16) unsigned short lds_h1[8][16][EDGE_H];    // 32 KB

    const int tid  = threadIdx.x;
    const int lane = tid & 31;
    const int wave = tid >> 5;
    const int h    = lane >> 4;   // K-half / M-half selector
    const int m    = lane & 15;   // A row within tile == C/D column n

    // Stage W2 fragments into LDS (reused 4*8 times by each of 8 waves).
    // Preferred path: CDNA5 async global->LDS DMA (ASYNCcnt), else sync copy.
#ifdef HAVE_ASYNC_LDS
    for (int i = tid; i < W2_FRAG_DW / 4; i += 256) {
        __builtin_amdgcn_global_load_async_to_lds_b128(
            (AS1 v4i*)(w2f + (size_t)i * 4),
            (AS3 v4i*)(lds_w2 + (size_t)i * 4),
            0, 0);
    }
    __builtin_amdgcn_s_wait_asynccnt(0);
#else
    {
        const uint4* s = (const uint4*)w2f;
        uint4* d = (uint4*)lds_w2;
        for (int i = tid; i < W2_FRAG_DW / 4; i += 256) d[i] = s[i];
    }
#endif
    __syncthreads();

    const int eBase = blockIdx.x * 128 + wave * 16;
    const int e_m   = eBase + m;
    const int srcN  = ei[e_m];
    const int dstN  = ei[E + e_m];
    const float* px_s = x  + (size_t)srcN * NODE_H;
    const float* px_d = x  + (size_t)dstN * NODE_H;
    const float* p_ea = ea + (size_t)e_m * EDGE_H;

    v8f acc[NT];
    for (int i = 0; i < NT; ++i)
        for (int r = 0; r < 8; ++r) acc[i][r] = 0.f;

    // ---------------- GEMM1: [16 x 384] x [384 x 128] ----------------
    for (int kt = 0; kt < KT1; ++kt) {
        // Build A fragment (ISA 7.12.2 16-bit A layout): lane holds row m,
        // elems 0..7 = cols kt*32 + h*8 .., elems 8..15 = +16. Each octet is
        // 8-col aligned -> never straddles the x[src]/x[dest]/ea boundaries.
        Frag a;
        const int c0 = kt * 32 + h * 8;
#pragma unroll
        for (int oc = 0; oc < 2; ++oc) {
            const int c = c0 + oc * 16;
            const float* p = (c < NODE_H)     ? (px_s + c)
                           : (c < 2 * NODE_H) ? (px_d + (c - NODE_H))
                                              : (p_ea + (c - 2 * NODE_H));
            float4 f0 = *(const float4*)(p);
            float4 f1 = *(const float4*)(p + 4);
            a.u[oc * 4 + 0] = pack_bf2(f0.x, f0.y);
            a.u[oc * 4 + 1] = pack_bf2(f0.z, f0.w);
            a.u[oc * 4 + 2] = pack_bf2(f1.x, f1.y);
            a.u[oc * 4 + 3] = pack_bf2(f1.z, f1.w);
        }
#pragma unroll
        for (int nt = 0; nt < NT; ++nt) {
            Frag b;
            const uint4* bp = (const uint4*)(w1f + ((kt * NT + nt) * 256 + lane * 8));
            b.q[0] = bp[0];
            b.q[1] = bp[1];
            acc[nt] = __builtin_amdgcn_wmma_f32_16x16x32_bf16(
                false, a.v, false, b.v, (short)0, acc[nt], false, false);
        }
    }

    // Epilogue 1: bias + leaky-relu (fp32), bf16 -> per-wave LDS tile.
    // D layout: elem r of acc[nt] = row 8h+r, col nt*16+m. Same-wave LDS
    // ops are in-order, so the GEMM2 reads below need no barrier.
#pragma unroll
    for (int nt = 0; nt < NT; ++nt) {
        const int col = nt * 16 + m;
        const float bias = b1[col];
#pragma unroll
        for (int r = 0; r < 8; ++r) {
            float v = acc[nt][r] + bias;
            v = (v > 0.f) ? v : v * 0.01f;
            lds_h1[wave][h * 8 + r][col] = f2bf_bits(v);
        }
    }

    // ---------------- GEMM2: [16 x 128] x [128 x 128] ----------------
    v8f acc2[NT];
    for (int i = 0; i < NT; ++i)
        for (int r = 0; r < 8; ++r) acc2[i][r] = 0.f;

    const unsigned short* hrow = lds_h1[wave][m];
    for (int kt = 0; kt < KT2; ++kt) {
        Frag a;
        const int c0 = kt * 32 + h * 8;           // ushort units; 16B aligned
        a.q[0] = *(const uint4*)(hrow + c0);      // ds_load_b128
        a.q[1] = *(const uint4*)(hrow + c0 + 16);
#pragma unroll
        for (int nt = 0; nt < NT; ++nt) {
            Frag b;
            const uint4* bp = (const uint4*)(lds_w2 + ((kt * NT + nt) * 256 + lane * 8));
            b.q[0] = bp[0];
            b.q[1] = bp[1];
            acc2[nt] = __builtin_amdgcn_wmma_f32_16x16x32_bf16(
                false, a.v, false, b.v, (short)0, acc2[nt], false, false);
        }
    }

    // Epilogue 2: bias + residual in fp32. Per (nt,r) a wave touches two
    // rows x 16 contiguous cols -> coalesced 64B segments; ea rows were just
    // read during the A-gather so these hit cache.
#pragma unroll
    for (int nt = 0; nt < NT; ++nt) {
        const int col = nt * 16 + m;
        const float bias = b2[col];
#pragma unroll
        for (int r = 0; r < 8; ++r) {
            const int e = eBase + h * 8 + r;
            const size_t idx = (size_t)e * EDGE_H + col;
            out[idx] = ea[idx] + acc2[nt][r] + bias;
        }
    }
}

// --------------------------------------------------------------------------
extern "C" void kernel_launch(void* const* d_in, const int* in_sizes, int n_in,
                              void* d_out, int out_size, void* d_ws, size_t ws_size,
                              hipStream_t stream) {
    const float* x  = (const float*)d_in[0];
    const int*   ei = (const int*)  d_in[1];   // edge_index (int32 under JAX defaults)
    const float* ea = (const float*)d_in[2];
    const float* W1 = (const float*)d_in[3];
    const float* b1 = (const float*)d_in[4];
    const float* W2 = (const float*)d_in[5];
    const float* b2 = (const float*)d_in[6];
    float* out = (float*)d_out;
    unsigned int* ws = (unsigned int*)d_ws;    // needs 128 KB

    const int E = in_sizes[1] / 2;             // 640000

    const int packThreads = W1_FRAG_DW + W2_FRAG_DW;
    pack_weights_kernel<<<(packThreads + 255) / 256, 256, 0, stream>>>(W1, W2, ws);

    edge_mlp_kernel<<<E / 128, 256, 0, stream>>>(
        x, ei, ea, b1, b2, ws, ws + W1_FRAG_DW, out, E);
}
